// iRPEMultiheadSelfAttention_47339129536648
// MI455X (gfx1250) — matrix operand order repaired and checked by reading, not compiled, not run on hardware
//
#include <hip/hip_runtime.h>

#define S_LEN    577
#define DMODEL   768
#define NHEAD    12
#define HDIM     64
#define NBUCKET  50
#define BATCH    16
#define NROWS    9232     /* B*S */
#define VT_STRIDE 608     /* padded j-stride for V^T: multiple of 32, covers j<608 */

typedef __attribute__((ext_vector_type(16))) __bf16    v16bf;
typedef __attribute__((ext_vector_type(4)))  __bf16    v4bf;
typedef __attribute__((ext_vector_type(8)))  float     v8f;
typedef __attribute__((ext_vector_type(4)))  float     f32x4;
typedef __attribute__((ext_vector_type(4)))  unsigned  u32x4;

union AF { v16bf v; u32x4 u[2]; };

static __device__ __forceinline__ float bf2f(__bf16 h) {
  unsigned u = ((unsigned)__builtin_bit_cast(unsigned short, h)) << 16;
  return __builtin_bit_cast(float, u);
}
static __device__ __forceinline__ v8f zero8() {
  v8f z;
  #pragma unroll
  for (int i = 0; i < 8; ++i) z[i] = 0.f;
  return z;
}
static __device__ __forceinline__ v8f wmma_bf16(v16bf a, v16bf b, v8f c) {
  // D = A(16x32) * B(32x16) + C, f32 accumulate
  return __builtin_amdgcn_wmma_f32_16x16x32_bf16(false, a, false, b, (short)0, c, false, false);
}

// ---------------------------------------------------------------------------
// Kernel 0: one-shot f32 -> bf16 conversion (native cvt), 4 elems/thread.
// ---------------------------------------------------------------------------
__global__ __launch_bounds__(256) void cvt_bf16(
    const float* __restrict__ src, __bf16* __restrict__ dst, int n4)
{
  const int i = blockIdx.x * 256 + threadIdx.x;
  if (i < n4) {
    f32x4 v = ((const f32x4*)src)[i];
    v4bf o;
    #pragma unroll
    for (int e = 0; e < 4; ++e) o[e] = (__bf16)v[e];
    ((v4bf*)dst)[i] = o;
  }
}

// ---------------------------------------------------------------------------
// Kernel 1: QKV = X @ W^T  (9232 x 2304, K=768), bf16 in / f32 accumulate.
// Wave computes 32x64; inner loop is pure b128 loads + 8 WMMAs per k-step.
// Stores Q (scaled 1/8) and K flat; V transposed per head (64x577, stride 608).
// grid (289, 9), block 128 (4 waves).
// ---------------------------------------------------------------------------
__global__ __launch_bounds__(128) void qkv_gemm(
    const __bf16* __restrict__ Xb, const __bf16* __restrict__ Wb,
    __bf16* __restrict__ Qb, __bf16* __restrict__ Kb, __bf16* __restrict__ Vt)
{
  const int lane = threadIdx.x & 31;
  const int wave = threadIdx.x >> 5;
  const int l16  = lane & 15;
  const int half = lane >> 4;
  const int mb   = blockIdx.x * 32;                 // row base (0..9216)
  const int nb   = blockIdx.y * 256 + wave * 64;    // col base

  int r0 = mb + l16;      if (r0 > NROWS - 1) r0 = NROWS - 1;
  int r1 = mb + 16 + l16; if (r1 > NROWS - 1) r1 = NROWS - 1;
  const u32x4* xr0 = (const u32x4*)(Xb + (size_t)r0 * DMODEL);  // 96 x 16B per row
  const u32x4* xr1 = (const u32x4*)(Xb + (size_t)r1 * DMODEL);
  const u32x4* wp[4];
  #pragma unroll
  for (int t = 0; t < 4; ++t)
    wp[t] = (const u32x4*)(Wb + (size_t)(nb + t * 16 + l16) * DMODEL);

  v8f acc0[4], acc1[4];
  #pragma unroll
  for (int t = 0; t < 4; ++t) { acc0[t] = zero8(); acc1[t] = zero8(); }

  for (int kc = 0; kc < 24; ++kc) {
    const int kb = kc * 4;                           // uint4 index of k-chunk
    AF a0, a1;                                       // A: K=(e/8)*16+half*8+e%8
    a0.u[0] = xr0[kb + half];  a0.u[1] = xr0[kb + 2 + half];
    a1.u[0] = xr1[kb + half];  a1.u[1] = xr1[kb + 2 + half];
    #pragma unroll
    for (int t = 0; t < 4; ++t) {
      AF bt;                                         // B: K=(e/8)*16+half*8+e%8
      bt.u[0] = wp[t][kb + half];
      bt.u[1] = wp[t][kb + 2 + half];
      acc0[t] = wmma_bf16(a0.v, bt.v, acc0[t]);
      acc1[t] = wmma_bf16(a1.v, bt.v, acc1[t]);
    }
  }

  const int sec = nb / DMODEL;                       // 0=q, 1=k, 2=v
  #pragma unroll
  for (int g = 0; g < 2; ++g) {
    #pragma unroll
    for (int t = 0; t < 4; ++t) {
      const int n = nb + t * 16 + l16;
      const int c = n - sec * DMODEL;
      #pragma unroll
      for (int r = 0; r < 8; ++r) {
        const int mg = mb + g * 16 + r + 8 * half;   // D frag: M = r + 8*half
        if (mg >= NROWS) continue;
        const float val = g ? acc1[t][r] : acc0[t][r];
        if (sec == 0) {
          Qb[(size_t)mg * DMODEL + c] = (__bf16)(val * 0.125f); // fold 1/sqrt(64)
        } else if (sec == 1) {
          Kb[(size_t)mg * DMODEL + c] = (__bf16)val;
        } else {
          // torch-style flat reshape: f = s*768 + c -> (h, s2, hd); store V^T
          const int bb  = mg / S_LEN;
          const int s   = mg - bb * S_LEN;
          const int f   = s * DMODEL + c;
          const int hh  = f / (S_LEN * HDIM);
          const int rem = f - hh * (S_LEN * HDIM);
          const int s2  = rem >> 6;
          const int hd  = rem & 63;
          Vt[((size_t)((bb * NHEAD + hh) * HDIM + hd)) * VT_STRIDE + s2] = (__bf16)val;
        }
      }
    }
  }
}

// ---------------------------------------------------------------------------
// Kernel 2: ctx[row, c] = Q_scaled[row,:] . rpe_table[:, c]   (row = bh*577+s2)
// ---------------------------------------------------------------------------
__global__ __launch_bounds__(256) void ctx_kernel(
    const __bf16* __restrict__ Qb, const float* __restrict__ rpe,
    float* __restrict__ ctx)
{
  __shared__ float Ts[HDIM * NBUCKET];
  __shared__ float qs[8][HDIM];
  for (int i = threadIdx.x; i < HDIM * NBUCKET; i += 256) Ts[i] = rpe[i];
  const int wave = threadIdx.x >> 5;
  const int lane = threadIdx.x & 31;
  const int row  = blockIdx.x * 8 + wave;            // exact: 110784 = 13848*8
  const __bf16* q = Qb + (size_t)row * HDIM;
  qs[wave][lane * 2]     = bf2f(q[lane * 2]);
  qs[wave][lane * 2 + 1] = bf2f(q[lane * 2 + 1]);
  __syncthreads();
  float a0 = 0.f, a1 = 0.f;
  const int c1 = lane + 32;
  #pragma unroll 4
  for (int d = 0; d < HDIM; ++d) {
    const float qv = qs[wave][d];
    a0 += qv * Ts[d * NBUCKET + lane];
    if (c1 < NBUCKET) a1 += qv * Ts[d * NBUCKET + c1];
  }
  float* out = ctx + (size_t)row * NBUCKET;
  out[lane] = a0;
  if (c1 < NBUCKET) out[c1] = a1;
}

// ---------------------------------------------------------------------------
// Kernel 3: fused flash attention, 32-row i-tile per wave (2 x 16 subtiles),
// 1 wave / block. K/V fragments shared across both subtiles; ctx rows staged
// in LDS so the per-(i,j) bias gather is a ds_load; rp_bucket loads stay
// coalesced VMEM. Online softmax with deferred (per-lane) row-sum.
// grid (19, 12, 16), block 32.
// ---------------------------------------------------------------------------
__global__ __launch_bounds__(32) void attn_kernel(
    const __bf16* __restrict__ Qb, const __bf16* __restrict__ Kb,
    const __bf16* __restrict__ Vt, const float* __restrict__ ctx,
    const int* __restrict__ rpb, float* __restrict__ y)
{
  __shared__ alignas(16) unsigned short Pl[2][16 * 32];
  __shared__ float Cl[32][52];                       // staged ctx rows (50 used)
  const int lane = threadIdx.x;
  const int l16  = lane & 15;
  const int half = lane >> 4;
  const int it = blockIdx.x, h = blockIdx.y, b = blockIdx.z;
  const int bh = b * NHEAD + h;
  const __bf16* Qh = Qb + (size_t)bh * S_LEN * HDIM;
  const __bf16* Kh = Kb + (size_t)bh * S_LEN * HDIM;
  const __bf16* Vh = Vt + (size_t)bh * HDIM * VT_STRIDE;
  const float* ctxh = ctx + (size_t)bh * S_LEN * NBUCKET;
  const int i0 = it * 32;

  // stage ctx rows i0..i0+31 (clamped) into LDS
  for (int idx = lane; idx < 32 * NBUCKET; idx += 32) {
    const int rr = idx / NBUCKET;
    const int cc = idx - rr * NBUCKET;
    int row = i0 + rr; if (row > S_LEN - 1) row = S_LEN - 1;
    Cl[rr][cc] = ctxh[(size_t)row * NBUCKET + cc];
  }

  AF a[2][2];                                        // [i-subtile][k-chunk]
  #pragma unroll
  for (int g = 0; g < 2; ++g) {
    int arow = i0 + g * 16 + l16; if (arow > S_LEN - 1) arow = S_LEN - 1;
    const u32x4* qp = (const u32x4*)(Qh + (size_t)arow * HDIM);
    a[g][0].u[0] = qp[half * 2];     a[g][0].u[1] = qp[half * 2 + 1];   // k 0..31
    a[g][1].u[0] = qp[4 + half * 2]; a[g][1].u[1] = qp[5 + half * 2];   // k 32..63
  }

  v8f o[2][4];
  float m8[2][8], l8[2][8];
  int qrow[2][8];
  #pragma unroll
  for (int g = 0; g < 2; ++g) {
    #pragma unroll
    for (int t = 0; t < 4; ++t) o[g][t] = zero8();
    #pragma unroll
    for (int r = 0; r < 8; ++r) {
      m8[g][r] = -__builtin_inff(); l8[g][r] = 0.f;
      const int mm = i0 + g * 16 + r + 8 * half;
      qrow[g][r] = mm > S_LEN - 1 ? S_LEN - 1 : mm;
    }
  }
  __syncthreads();                                   // Cl ready

  for (int jt = 0; jt < 19; ++jt) {
    const int j0  = jt * 32;
    const int jr0 = j0 + l16, jr1 = j0 + 16 + l16;
    const int jc0 = jr0 > S_LEN - 1 ? S_LEN - 1 : jr0;
    const int jc1 = jr1 > S_LEN - 1 ? S_LEN - 1 : jr1;
    if (jt < 18)
      __builtin_prefetch(Kh + (size_t)(jc0 + 32 > S_LEN - 1 ? S_LEN - 1 : jc0 + 32) * HDIM, 0, 1);

    const u32x4* kp0 = (const u32x4*)(Kh + (size_t)jc0 * HDIM);
    const u32x4* kp1 = (const u32x4*)(Kh + (size_t)jc1 * HDIM);
    AF bk[2][2];                                     // [j-subtile][k-chunk]
    bk[0][0].u[0] = kp0[half * 2];     bk[0][0].u[1] = kp0[half * 2 + 1];
    bk[0][1].u[0] = kp0[4 + half * 2]; bk[0][1].u[1] = kp0[4 + half * 2 + 1];
    bk[1][0].u[0] = kp1[half * 2];     bk[1][0].u[1] = kp1[half * 2 + 1];
    bk[1][1].u[0] = kp1[4 + half * 2]; bk[1][1].u[1] = kp1[4 + half * 2 + 1];

    v8f s[2][2];                                     // [i-subtile][j-subtile]
    #pragma unroll
    for (int g = 0; g < 2; ++g) {
      #pragma unroll
      for (int jf = 0; jf < 2; ++jf) {
        v8f acc = zero8();
        acc = wmma_bf16(a[g][0].v, bk[jf][0].v, acc);
        acc = wmma_bf16(a[g][1].v, bk[jf][1].v, acc);
        s[g][jf] = acc;
      }
    }

    // iRPE contextual bias from LDS-staged ctx; mask j >= 577
    #pragma unroll
    for (int g = 0; g < 2; ++g) {
      #pragma unroll
      for (int r = 0; r < 8; ++r) {
        const int qr = qrow[g][r];
        const int lrow = g * 16 + r + 8 * half;
        if (jr0 < S_LEN) s[g][0][r] += Cl[lrow][rpb[qr * S_LEN + jr0]];
        else             s[g][0][r]  = -__builtin_inff();
        if (jr1 < S_LEN) s[g][1][r] += Cl[lrow][rpb[qr * S_LEN + jr1]];
        else             s[g][1][r]  = -__builtin_inff();
      }
    }

    // online softmax; l kept as per-lane partial (alpha is row-uniform)
    #pragma unroll
    for (int g = 0; g < 2; ++g) {
      #pragma unroll
      for (int r = 0; r < 8; ++r) {
        float v = fmaxf(s[g][0][r], s[g][1][r]);
        v = fmaxf(v, __shfl_xor(v, 1, 32));
        v = fmaxf(v, __shfl_xor(v, 2, 32));
        v = fmaxf(v, __shfl_xor(v, 4, 32));
        v = fmaxf(v, __shfl_xor(v, 8, 32));
        const float mn = fmaxf(m8[g][r], v);
        const float al = __expf(m8[g][r] - mn);
        const float p0 = __expf(s[g][0][r] - mn);
        const float p1 = __expf(s[g][1][r] - mn);
        l8[g][r] = l8[g][r] * al + (p0 + p1);
        m8[g][r] = mn;
        #pragma unroll
        for (int t = 0; t < 4; ++t) o[g][t][r] *= al;
        const int mm = r + 8 * half;                 // stage P (D-layout -> LDS)
        Pl[g][mm * 32 + l16]      = __builtin_bit_cast(unsigned short, (__bf16)p0);
        Pl[g][mm * 32 + 16 + l16] = __builtin_bit_cast(unsigned short, (__bf16)p1);
      }
    }
    __syncthreads();

    AF pa[2];                                        // P as A frag (16x32)
    #pragma unroll
    for (int g = 0; g < 2; ++g) {
      const u32x4* pp = (const u32x4*)Pl[g] + l16 * 4;
      pa[g].u[0] = pp[half * 2];
      pa[g].u[1] = pp[half * 2 + 1];
    }
    #pragma unroll
    for (int t = 0; t < 4; ++t) {                    // V frag shared across g
      const int drow = t * 16 + l16;                 // V^T row d: contiguous in j
      const u32x4* vp = (const u32x4*)(Vh + (size_t)drow * VT_STRIDE + j0 + half * 16);
      AF bv; bv.u[0] = vp[0]; bv.u[1] = vp[1];
      o[0][t] = wmma_bf16(pa[0].v, bv.v, o[0][t]);
      o[1][t] = wmma_bf16(pa[1].v, bv.v, o[1][t]);
    }
    __syncthreads();
  }

  // final row-sum reduce + normalize; out.transpose: y[b, s2, h*64 + d]
  #pragma unroll
  for (int g = 0; g < 2; ++g) {
    #pragma unroll
    for (int r = 0; r < 8; ++r) {
      float ls = l8[g][r];
      ls += __shfl_xor(ls, 1, 32);
      ls += __shfl_xor(ls, 2, 32);
      ls += __shfl_xor(ls, 4, 32);
      ls += __shfl_xor(ls, 8, 32);
      const int mm = i0 + g * 16 + r + 8 * half;
      if (mm < S_LEN) {
        const float inv = 1.0f / ls;
        float* yp = y + ((size_t)b * S_LEN + mm) * DMODEL + h * HDIM;
        #pragma unroll
        for (int t = 0; t < 4; ++t) yp[t * 16 + l16] = o[g][t][r] * inv;
      }
    }
  }
}

// ---------------------------------------------------------------------------
extern "C" void kernel_launch(void* const* d_in, const int* in_sizes, int n_in,
                              void* d_out, int out_size, void* d_ws, size_t ws_size,
                              hipStream_t stream) {
  (void)in_sizes; (void)n_in; (void)out_size; (void)ws_size;
  const float* x   = (const float*)d_in[0];   // (B,S,D) f32
  const float* w   = (const float*)d_in[1];   // (3D,D)  f32
  const float* rpe = (const float*)d_in[2];   // (64,50) f32
  const int*   rpb = (const int*)d_in[3];     // (577,577) i32
  float* y = (float*)d_out;                   // (B,S,D) f32
  char* ws = (char*)d_ws;
  // workspace layout (bytes)
  __bf16* Qb  = (__bf16*)(ws + 0);            // 14,180,352
  __bf16* Kb  = (__bf16*)(ws + 14180352);     // 14,180,352
  __bf16* Vt  = (__bf16*)(ws + 28360704);     // 14,942,208
  float*  ctx = (float*)(ws + 43302912);      // 22,156,800
  __bf16* Xb  = (__bf16*)(ws + 65459712);     // 14,180,352
  __bf16* Wb  = (__bf16*)(ws + 79640064);     // 3,538,944  -> total ~83.2 MB

  cvt_bf16   <<<dim3(6924), 256, 0, stream>>>(x, Xb, 1772544);   // 9232*768/4
  cvt_bf16   <<<dim3(1728), 256, 0, stream>>>(w, Wb, 442368);    // 2304*768/4
  qkv_gemm   <<<dim3(289, 9),    128, 0, stream>>>(Xb, Wb, Qb, Kb, Vt);
  ctx_kernel <<<dim3(13848),     256, 0, stream>>>(Qb, rpe, ctx);
  attn_kernel<<<dim3(19, 12, 16), 32, 0, stream>>>(Qb, Kb, Vt, ctx, rpb, y);
}
